// GAT_Encoder_60129542496
// MI455X (gfx1250) — matrix operand order repaired and checked by reading, not compiled, not run on hardware
//
#include <hip/hip_runtime.h>
#include <hip/hip_bf16.h>

typedef __attribute__((ext_vector_type(16))) __bf16 v16bf;
typedef __attribute__((ext_vector_type(8)))  float  v8f;

#define B_    8
#define NV_   1024
#define FEAT_ 256
#define NH_   4
#define DH_   64
#define NBLK_ 2
#define ALPHA_ 0.2f
#define EPS_   1e-5f
#define NEGF_  -9.0e15f

// ---------------------------------------------------------------------------
// BatchNorm1d(nvars) on (B, NV, FEAT): channel = NV axis, stats over (B, FEAT)
// ---------------------------------------------------------------------------
__global__ __launch_bounds__(256) void bn_kernel(const float* __restrict__ src,
                                                 float* __restrict__ dst,
                                                 const float* __restrict__ gamma,
                                                 const float* __restrict__ beta) {
    const int v = blockIdx.x;
    const int t = threadIdx.x;
    __shared__ float s_sum[256];
    __shared__ float s_sq[256];

    float sum = 0.f, sq = 0.f;
#pragma unroll
    for (int b = 0; b < B_; ++b) {
        float x = src[((size_t)b * NV_ + v) * FEAT_ + t];
        sum += x; sq += x * x;
    }
    s_sum[t] = sum; s_sq[t] = sq;
    __syncthreads();
    for (int off = 128; off > 0; off >>= 1) {
        if (t < off) { s_sum[t] += s_sum[t + off]; s_sq[t] += s_sq[t + off]; }
        __syncthreads();
    }
    const float n    = (float)(B_ * FEAT_);
    const float mean = s_sum[0] / n;
    const float var  = s_sq[0] / n - mean * mean;       // biased, torch training fwd
    const float inv  = rsqrtf(var + EPS_);
    const float g = gamma[v], be = beta[v];
#pragma unroll
    for (int b = 0; b < B_; ++b) {
        float x = src[((size_t)b * NV_ + v) * FEAT_ + t];
        dst[((size_t)b * NV_ + v) * FEAT_ + t] = g * (x - mean) * inv + be;
    }
}

// ---------------------------------------------------------------------------
// Pre-swizzle W (fp32) into bf16 WMMA-B fragment layout:
//   Wswz[(((h*(FEAT/32) + kc)*2 + hi)*DH + d)*16 + i] = bf16(W[h][kc*32+hi*16+i][d])
// A lane's B fragment (16 bf16) is then 32 contiguous bytes.
// ---------------------------------------------------------------------------
__global__ __launch_bounds__(256) void wswz_kernel(const float* __restrict__ W,
                                                   __bf16* __restrict__ Wswz) {
    int idx = blockIdx.x * 256 + threadIdx.x;     // over NH*FEAT*DH
    int h   = idx / (FEAT_ * DH_);
    int rem = idx - h * (FEAT_ * DH_);
    int k   = rem / DH_;
    int d   = rem - k * DH_;
    int kc  = k >> 5;
    int hi  = (k >> 4) & 1;
    int i   = k & 15;
    Wswz[((size_t)(((h * (FEAT_ / 32) + kc) * 2 + hi) * DH_) + d) * 16 + i] = (__bf16)W[idx];
}

// ---------------------------------------------------------------------------
// Wh[b,h,n,d] = sum_f enc[b,n,f] * W[h,f,d] + Wb[h,d]
// Grid: B*NH*(NV/16) blocks, 128 threads = 4 waves; wave w owns d-tile w.
// A tile staged to LDS as bf16 cooperatively (shared by 4 waves); B fragment
// from pre-swizzled Wswz (2x b128). Epilogue stores Wh fp32 AND the bf16
// fragment-layout copy Whb (one b128 store per lane) for the attn kernel.
// ---------------------------------------------------------------------------
__global__ __launch_bounds__(128) void feat_gemm_kernel(const float* __restrict__ enc,
                                                        const __bf16* __restrict__ Wswz,
                                                        const float* __restrict__ Wb,
                                                        float* __restrict__ Wh,
                                                        __bf16* __restrict__ Whb) {
    const int bid  = blockIdx.x;
    const int tile = bid & 63;          // NV/16 = 64 tiles
    const int bh   = bid >> 6;          // b*NH + h
    const int h    = bh & (NH_ - 1);
    const int b    = bh >> 2;
    const int n0   = tile * 16;
    const int t    = threadIdx.x;
    const int wave = t >> 5;
    const int lane = t & 31;
    const int d0   = wave * 16;
    const int mL   = lane & 15;
    const int hi   = lane >> 4;
    const int kbA  = hi ? 16 : 0;       // A-frag K base (contiguous k = 16*hi + s)

    __shared__ __align__(16) __bf16 As[16][40];   // 16 rows x 32 K (+8 pad)

    const int ti = t & 15;              // cooperative-load row
    const int jg = t >> 4;              // 0..7, 4 K values each
    const float*  __restrict__ aload = enc + ((size_t)b * NV_ + n0 + ti) * FEAT_ + jg * 4;
    const __bf16* __restrict__ bbase = Wswz
        + ((size_t)(h * (FEAT_ / 32)) * 2 + hi) * DH_ * 16 + (size_t)(d0 + mL) * 16;

    v8f acc = {};
    for (int k0 = 0; k0 < FEAT_; k0 += 32) {
        // --- cooperative fp32 -> bf16 A-tile stage (each thread: 4 elements)
        float4 f = *(const float4*)(aload + k0);
        union { __bf16 h4[4]; uint2 u; } pk;
        pk.h4[0] = (__bf16)f.x; pk.h4[1] = (__bf16)f.y;
        pk.h4[2] = (__bf16)f.z; pk.h4[3] = (__bf16)f.w;
        *(uint2*)&As[ti][jg * 4] = pk.u;
        __syncthreads();

        // --- A fragment: two ds_load_b128 from this lane's row
        union { v16bf v; uint4 q[2]; } fa;
        fa.q[0] = *(const uint4*)&As[mL][kbA];
        fa.q[1] = *(const uint4*)&As[mL][kbA + 8];

        // --- B fragment: two coalesced b128 loads from swizzled bf16 W
        union { v16bf v; uint4 q[2]; } fb;
        const __bf16* bp = bbase + (size_t)(k0 >> 5) * 2 * DH_ * 16;
        fb.q[0] = *(const uint4*)(bp);
        fb.q[1] = *(const uint4*)(bp + 8);

        acc = __builtin_amdgcn_wmma_f32_16x16x32_bf16(false, fa.v, false, fb.v,
                                                      (short)0, acc, false, false);
        __syncthreads();
    }
    const float bias = Wb[h * DH_ + d0 + mL];

    // fp32 Wh (for score vectors)
#pragma unroll
    for (int r = 0; r < 8; ++r) {
        int mm = r + 8 * hi;            // C/D layout: VGPR r -> row r (+8 for hi lanes)
        Wh[((size_t)bh * NV_ + n0 + mm) * DH_ + d0 + mL] = acc[r] + bias;
    }
    // bf16 Whb in WMMA-B fragment layout: this lane's 8 values are one
    // contiguous i-run -> single b128 store.
    union { __bf16 h8[8]; uint4 q; } pw;
#pragma unroll
    for (int r = 0; r < 8; ++r) pw.h8[r] = (__bf16)(acc[r] + bias);
    size_t woff = ((((size_t)bh * (NV_ / 32) + (n0 >> 5)) * 2 + ((n0 >> 4) & 1)) * DH_
                   + (d0 + mL)) * 16 + 8 * hi;
    *(uint4*)(Whb + woff) = pw.q;
}

// ---------------------------------------------------------------------------
// Wh1[bh,n] = Wh[bh,n,:] . a1[h,:]   ;   Wh2 likewise with a2
// ---------------------------------------------------------------------------
__global__ __launch_bounds__(256) void score_kernel(const float* __restrict__ Wh,
                                                    const float* __restrict__ a1,
                                                    const float* __restrict__ a2,
                                                    float* __restrict__ Wh1,
                                                    float* __restrict__ Wh2) {
    int idx = blockIdx.x * 256 + threadIdx.x;
    if (idx >= B_ * NH_ * NV_) return;
    int bh = idx / NV_;
    int h  = bh & (NH_ - 1);
    const float* __restrict__ row = Wh + (size_t)idx * DH_;
    float s1 = 0.f, s2 = 0.f;
#pragma unroll
    for (int d = 0; d < DH_; ++d) {
        float w = row[d];
        s1 += w * a1[h * DH_ + d];
        s2 += w * a2[h * DH_ + d];
    }
    Wh1[idx] = s1;
    Wh2[idx] = s2;
}

// ---------------------------------------------------------------------------
// Fused GAT attention (flash style; rank-1 scores so no QK matmul needed):
//   s[j,i] = lrelu(Wh1[j] + Wh2[i]) masked by adj[b,j,i]; softmax over j;
//   hp[i,d] = sum_j p[j] * Wh[j,d]; ELU.
// Two-pass softmax (adj L2-resident). Pass 2 inner loop per wave:
//   2x ds_load_b128 (P from LDS) + 2x global b128 (V from swizzled Whb) + WMMA.
// ---------------------------------------------------------------------------
__global__ __launch_bounds__(128) void attn_kernel(const __bf16* __restrict__ Whb,
                                                   const float* __restrict__ Wh1,
                                                   const float* __restrict__ Wh2,
                                                   const int* __restrict__ adj,
                                                   float* __restrict__ out) {
    const int bid  = blockIdx.x;
    const int tile = bid & 63;
    const int bh   = bid >> 6;
    const int h    = bh & (NH_ - 1);
    const int b    = bh >> 2;
    const int i0   = tile * 16;
    const int t    = threadIdx.x;

    __shared__ float s_red[128];
    __shared__ float s_rowmax[16];
    __shared__ float s_rowsum[16];
    __shared__ __align__(16) __bf16 Ps[16][40];   // 16 i-rows x 32 K (+8 pad)

    const int ti = t & 15;
    const int g  = t >> 4;              // 8 j-strided groups
    const float w2 = Wh2[(size_t)bh * NV_ + i0 + ti];

    // ---- pass 1a: row max over j
    float mx = -3.0e38f;
    for (int j = g; j < NV_; j += 8) {
        int   ad = adj[((size_t)b * NV_ + j) * NV_ + i0 + ti];
        float e  = Wh1[(size_t)bh * NV_ + j] + w2;
        e = (e >= 0.f) ? e : ALPHA_ * e;
        float s = (ad > 0) ? e : NEGF_;
        mx = fmaxf(mx, s);
    }
    s_red[t] = mx;
    __syncthreads();
    if (g == 0) {
        float v = s_red[ti];
#pragma unroll
        for (int q = 1; q < 8; ++q) v = fmaxf(v, s_red[q * 16 + ti]);
        s_rowmax[ti] = v;
    }
    __syncthreads();

    // ---- pass 1b: row sum of exp
    const float rmx = s_rowmax[ti];
    float sm = 0.f;
    for (int j = g; j < NV_; j += 8) {
        int   ad = adj[((size_t)b * NV_ + j) * NV_ + i0 + ti];
        float e  = Wh1[(size_t)bh * NV_ + j] + w2;
        e = (e >= 0.f) ? e : ALPHA_ * e;
        float s = (ad > 0) ? e : NEGF_;
        sm += __expf(s - rmx);
    }
    s_red[t] = sm;
    __syncthreads();
    if (g == 0) {
        float v = s_red[ti];
#pragma unroll
        for (int q = 1; q < 8; ++q) v += s_red[q * 16 + ti];
        s_rowsum[ti] = v;
    }
    __syncthreads();

    // ---- pass 2: WMMA accumulate P (16 x 32) x Wh (32 x 16) over j chunks
    const int wave = t >> 5;
    const int lane = t & 31;
    const int d0   = wave * 16;
    const int mL   = lane & 15;
    const int hi   = lane >> 4;
    const int kbA  = hi ? 16 : 0;

    const __bf16* __restrict__ vbase = Whb
        + ((size_t)bh * (NV_ / 32) * 2 + hi) * DH_ * 16 + (size_t)(d0 + mL) * 16;

    v8f acc = {};
    for (int j0 = 0; j0 < NV_; j0 += 32) {
        // --- cooperative P-tile build: thread (ti, g) covers j = j0+g*4+[0,4)
        float4 w1 = *(const float4*)(Wh1 + (size_t)bh * NV_ + j0 + g * 4);
        union { __bf16 h4[4]; uint2 u; } pk;
#pragma unroll
        for (int q = 0; q < 4; ++q) {
            int j = j0 + g * 4 + q;
            int   ad = adj[((size_t)b * NV_ + j) * NV_ + i0 + ti];
            float e  = ((const float*)&w1)[q] + w2;
            e = (e >= 0.f) ? e : ALPHA_ * e;
            float s = (ad > 0) ? e : NEGF_;
            pk.h4[q] = (__bf16)__expf(s - rmx);
        }
        *(uint2*)&Ps[ti][g * 4] = pk.u;
        __syncthreads();

        // --- A fragment from LDS (two b128 loads of this lane's i-row)
        union { v16bf v; uint4 q2[2]; } fa;
        fa.q2[0] = *(const uint4*)&Ps[mL][kbA];
        fa.q2[1] = *(const uint4*)&Ps[mL][kbA + 8];

        // --- V fragment: two coalesced b128 loads from swizzled bf16 Whb
        union { v16bf v; uint4 q2[2]; } fb;
        const __bf16* vp = vbase + (size_t)(j0 >> 5) * 2 * DH_ * 16;
        fb.q2[0] = *(const uint4*)(vp);
        fb.q2[1] = *(const uint4*)(vp + 8);

        acc = __builtin_amdgcn_wmma_f32_16x16x32_bf16(false, fa.v, false, fb.v,
                                                      (short)0, acc, false, false);
        __syncthreads();
    }
#pragma unroll
    for (int r = 0; r < 8; ++r) {
        int mm = r + 8 * hi;
        float o = acc[r] / s_rowsum[mm];
        o = (o > 0.f) ? o : (__expf(o) - 1.f);          // ELU
        // concat heads: out[b, i, h*DH + d]
        out[((size_t)b * NV_ + i0 + mm) * FEAT_ + h * DH_ + d0 + mL] = o;
    }
}

// ---------------------------------------------------------------------------
extern "C" void kernel_launch(void* const* d_in, const int* in_sizes, int n_in,
                              void* d_out, int out_size, void* d_ws, size_t ws_size,
                              hipStream_t stream) {
    const float* x      = (const float*)d_in[0];
    const int*   adj    = (const int*)  d_in[1];
    const float* W      = (const float*)d_in[2];
    const float* Wb     = (const float*)d_in[3];
    const float* a1     = (const float*)d_in[4];
    const float* a2     = (const float*)d_in[5];
    const float* gamma0 = (const float*)d_in[6];
    const float* beta0  = (const float*)d_in[7];
    const float* gammas = (const float*)d_in[8];
    const float* betas  = (const float*)d_in[9];

    float* ws  = (float*)d_ws;
    float* enc = ws;                                     // 2M floats (8MB)
    float* tmp = ws + (size_t)2 * 1024 * 1024;           // 2M floats
    float* Wh  = ws + (size_t)4 * 1024 * 1024;           // 2M floats
    float* Wh1 = ws + (size_t)6 * 1024 * 1024;           // 32K floats
    float* Wh2 = Wh1 + (size_t)B_ * NH_ * NV_;           // 32K floats
    __bf16* Wswz = (__bf16*)(ws + (size_t)6 * 1024 * 1024 + 64 * 1024);   // 64K bf16
    __bf16* Whb  = (__bf16*)(ws + (size_t)6 * 1024 * 1024 + 128 * 1024);  // 2M bf16 (4MB)

    bn_kernel<<<NV_, 256, 0, stream>>>(x, enc, gamma0, beta0);

    for (int blk = 0; blk < NBLK_; ++blk) {
        const float* Wp  = W  + (size_t)blk * NH_ * FEAT_ * DH_;
        const float* Wbp = Wb + (size_t)blk * NH_ * DH_;
        const float* a1p = a1 + (size_t)blk * NH_ * DH_;
        const float* a2p = a2 + (size_t)blk * NH_ * DH_;

        wswz_kernel<<<(NH_ * FEAT_ * DH_) / 256, 256, 0, stream>>>(Wp, Wswz);
        feat_gemm_kernel<<<B_ * NH_ * (NV_ / 16), 128, 0, stream>>>(enc, Wswz, Wbp, Wh, Whb);
        score_kernel<<<(B_ * NH_ * NV_ + 255) / 256, 256, 0, stream>>>(Wh, a1p, a2p, Wh1, Wh2);
        attn_kernel<<<B_ * NH_ * (NV_ / 16), 128, 0, stream>>>(Whb, Wh1, Wh2, adj, tmp);

        float* dst = (blk == NBLK_ - 1) ? (float*)d_out : enc;
        bn_kernel<<<NV_, 256, 0, stream>>>(tmp, dst,
                                           gammas + (size_t)blk * NV_,
                                           betas  + (size_t)blk * NV_);
    }
}
